// RwkvBitSelfAttention_50740743635386
// MI455X (gfx1250) — 8-bit matrix operand order repaired and checked by reading, compiled, not run on hardware
//
#include <hip/hip_runtime.h>
#include <hip/hip_bf16.h>
#include <stdint.h>


#define Bz 8
#define Tz 2048
#define Hz 1024
#define Dz 1024
#define Mtok (Bz * Tz)
#define Kdim 1024
#define Ndim 1024

#define NCH   (Bz * Dz)
#define CHK   32
#define CHL   (Tz / CHK)

typedef __attribute__((ext_vector_type(8))) int v8i;

__device__ __forceinline__ float blockReduceSum256(float v, float* red) {
    int tid = threadIdx.x;
    red[tid] = v;
    __syncthreads();
    for (int s = 128; s > 0; s >>= 1) {
        if (tid < s) red[tid] += red[tid + s];
        __syncthreads();
    }
    float r = red[0];
    __syncthreads();
    return r;
}

__device__ __forceinline__ float blockReduceMax256(float v, float* red) {
    int tid = threadIdx.x;
    red[tid] = v;
    __syncthreads();
    for (int s = 128; s > 0; s >>= 1) {
        if (tid < s) red[tid] = fmaxf(red[tid], red[tid + s]);
        __syncthreads();
    }
    float r = red[0];
    __syncthreads();
    return r;
}

__global__ void __launch_bounds__(256)
w_abssum(const float* __restrict__ w0, const float* __restrict__ w1,
         const float* __restrict__ w2, const float* __restrict__ w3,
         float* __restrict__ partials) {
    const float* ws[4] = {w0, w1, w2, w3};
    const float* w = ws[blockIdx.y];
    __shared__ float red[256];
    float s = 0.f;
    for (int i = blockIdx.x * 256 + threadIdx.x; i < 262144; i += gridDim.x * 256) {
        float4 v = *(const float4*)(w + (size_t)i * 4);
        s += fabsf(v.x) + fabsf(v.y) + fabsf(v.z) + fabsf(v.w);
    }
    s = blockReduceSum256(s, red);
    if (threadIdx.x == 0) partials[blockIdx.y * 64 + blockIdx.x] = s;
}

__global__ void __launch_bounds__(64)
w_finalize(const float* __restrict__ partials, float* __restrict__ wdeq) {
    __shared__ float red[64];
    int tid = threadIdx.x;
    red[tid] = partials[blockIdx.x * 64 + tid];
    __syncthreads();
    for (int s = 32; s > 0; s >>= 1) {
        if (tid < s) red[tid] += red[tid + s];
        __syncthreads();
    }
    if (tid == 0) {
        float mean = red[0] * (1.0f / (1024.0f * 1024.0f));
        wdeq[blockIdx.x] = fmaxf(mean, 1e-5f);
    }
}

__global__ void __launch_bounds__(256)
w_quant(const float* __restrict__ w0, const float* __restrict__ w1,
        const float* __restrict__ w2, const float* __restrict__ w3,
        const float* __restrict__ wdeq,
        signed char* __restrict__ q0, signed char* __restrict__ q1,
        signed char* __restrict__ q2, signed char* __restrict__ q3) {
    const float* ws[4] = {w0, w1, w2, w3};
    signed char* qs[4] = {q0, q1, q2, q3};
    int wi = blockIdx.y;
    const float* w = ws[wi];
    signed char* q = qs[wi];
    float scale = 1.0f / wdeq[wi];
    int i = blockIdx.x * 256 + threadIdx.x;
    float4 v = *(const float4*)(w + (size_t)i * 4);
    char4 c;
    c.x = (signed char)fminf(fmaxf(rintf(v.x * scale), -1.f), 1.f);
    c.y = (signed char)fminf(fmaxf(rintf(v.y * scale), -1.f), 1.f);
    c.z = (signed char)fminf(fmaxf(rintf(v.z * scale), -1.f), 1.f);
    c.w = (signed char)fminf(fmaxf(rintf(v.w * scale), -1.f), 1.f);
    *(char4*)(q + (size_t)i * 4) = c;
}

__global__ void __launch_bounds__(256)
mix_rms_quant(const float* __restrict__ hidden,
              const float* __restrict__ tmk, const float* __restrict__ tmv,
              const float* __restrict__ tmr,
              const float* __restrict__ rmsk, const float* __restrict__ rmsv,
              const float* __restrict__ rmsr,
              signed char* __restrict__ qk, signed char* __restrict__ qv,
              signed char* __restrict__ qr,
              float* __restrict__ rdk, float* __restrict__ rdv,
              float* __restrict__ rdr) {
    __shared__ float red[256];
    int token = blockIdx.x;
    int t = token & (Tz - 1);
    int tid = threadIdx.x;
    size_t base = (size_t)token * Hz + tid * 4;

    float4 h = *(const float4*)(hidden + base);
    float4 s = make_float4(0.f, 0.f, 0.f, 0.f);
    if (t > 0) s = *(const float4*)(hidden + base - Hz);

    const float* tms[3] = {tmk, tmv, tmr};
    const float* rms[3] = {rmsk, rmsv, rmsr};
    signed char* qs[3]  = {qk, qv, qr};
    float* rds[3]       = {rdk, rdv, rdr};

    for (int mix = 0; mix < 3; mix++) {
        float4 tm = *(const float4*)(tms[mix] + tid * 4);
        float4 x;
        x.x = h.x * tm.x + s.x * (1.f - tm.x);
        x.y = h.y * tm.y + s.y * (1.f - tm.y);
        x.z = h.z * tm.z + s.z * (1.f - tm.z);
        x.w = h.w * tm.w + s.w * (1.f - tm.w);

        float ssq = x.x * x.x + x.y * x.y + x.z * x.z + x.w * x.w;
        float tot = blockReduceSum256(ssq, red);
        float rinv = rsqrtf(tot * (1.0f / (float)Hz) + 1e-8f);

        float4 g = *(const float4*)(rms[mix] + tid * 4);
        float4 xn;
        xn.x = x.x * rinv * g.x;
        xn.y = x.y * rinv * g.y;
        xn.z = x.z * rinv * g.z;
        xn.w = x.w * rinv * g.w;

        float am = fmaxf(fmaxf(fabsf(xn.x), fabsf(xn.y)),
                         fmaxf(fabsf(xn.z), fabsf(xn.w)));
        am = blockReduceMax256(am, red);
        float scale = 127.0f / fmaxf(am, 1e-5f);

        char4 q;
        q.x = (signed char)fminf(fmaxf(rintf(xn.x * scale), -128.f), 127.f);
        q.y = (signed char)fminf(fmaxf(rintf(xn.y * scale), -128.f), 127.f);
        q.z = (signed char)fminf(fmaxf(rintf(xn.z * scale), -128.f), 127.f);
        q.w = (signed char)fminf(fmaxf(rintf(xn.w * scale), -128.f), 127.f);
        *(char4*)(qs[mix] + base) = q;
        if (tid == 0) rds[mix][token] = 1.0f / scale;
    }
}

__global__ void __launch_bounds__(256)
rms_quant_single(const float* __restrict__ y, const float* __restrict__ rmsw,
                 signed char* __restrict__ q, float* __restrict__ rd) {
    __shared__ float red[256];
    int token = blockIdx.x;
    int tid = threadIdx.x;
    size_t base = (size_t)token * Dz + tid * 4;

    float4 x = *(const float4*)(y + base);
    float ssq = x.x * x.x + x.y * x.y + x.z * x.z + x.w * x.w;
    float tot = blockReduceSum256(ssq, red);
    float rinv = rsqrtf(tot * (1.0f / (float)Dz) + 1e-8f);

    float4 g = *(const float4*)(rmsw + tid * 4);
    float4 xn;
    xn.x = x.x * rinv * g.x;
    xn.y = x.y * rinv * g.y;
    xn.z = x.z * rinv * g.z;
    xn.w = x.w * rinv * g.w;

    float am = fmaxf(fmaxf(fabsf(xn.x), fabsf(xn.y)),
                     fmaxf(fabsf(xn.z), fabsf(xn.w)));
    am = blockReduceMax256(am, red);
    float scale = 127.0f / fmaxf(am, 1e-5f);

    char4 qc;
    qc.x = (signed char)fminf(fmaxf(rintf(xn.x * scale), -128.f), 127.f);
    qc.y = (signed char)fminf(fmaxf(rintf(xn.y * scale), -128.f), 127.f);
    qc.z = (signed char)fminf(fmaxf(rintf(xn.z * scale), -128.f), 127.f);
    qc.w = (signed char)fminf(fmaxf(rintf(xn.w * scale), -128.f), 127.f);
    *(char4*)(q + base) = qc;
    if (tid == 0) rd[token] = 1.0f / scale;
}

__device__ __forceinline__ void async_b128_to_lds(const signed char* gaddr,
                                                  unsigned lds_addr) {
    asm volatile("global_load_async_to_lds_b128 %0, %1, off"
                 :: "v"(lds_addr), "v"(gaddr) : "memory");
}

__device__ __forceinline__ void wait_asynccnt0() {
    asm volatile("s_wait_asynccnt 0x0" ::: "memory");
}

__global__ void __launch_bounds__(256)
gemm_i8(const signed char* __restrict__ A, const signed char* __restrict__ Bw,
        const float* __restrict__ row_deq, const float* __restrict__ wdeq_ptr,
        float* __restrict__ C, int mode) {
    __shared__ __align__(16) signed char lsA[2][256 * 64];
    __shared__ __align__(16) signed char lsB[2][64 * 64];

    int tid  = threadIdx.x;
    int m0   = blockIdx.x * 256;
    int n0   = blockIdx.y * 64;
    int wave = tid >> 5;
    int lane = tid & 31;
    int m_l  = lane & 15;
    int half = lane >> 4;

    v8i acc[2][4];
#pragma unroll
    for (int s = 0; s < 2; s++)
#pragma unroll
        for (int i = 0; i < 4; i++) acc[s][i] = (v8i){0, 0, 0, 0, 0, 0, 0, 0};

    int srow = tid >> 2;
    int soff = (tid & 3) * 16;

    const signed char* gA = A + (size_t)(m0 + srow) * Kdim + soff;
    const signed char* gB = Bw + (size_t)(n0 + srow) * Kdim + soff;

    auto issue = [&](int buf, int kc) {
#pragma unroll
        for (int c = 0; c < 4; c++) {
            unsigned la = (unsigned)(uintptr_t)&lsA[buf][(srow + c * 64) * 64 + soff];
            async_b128_to_lds(gA + (size_t)c * 64 * Kdim + kc, la);
        }
        unsigned lb = (unsigned)(uintptr_t)&lsB[buf][srow * 64 + soff];
        async_b128_to_lds(gB + kc, lb);
    };

    issue(0, 0);

    for (int i = 0; i < Kdim / 64; i++) {
        int cur = i & 1;
        wait_asynccnt0();
        __syncthreads();
        if (i + 1 < Kdim / 64) issue(cur ^ 1, (i + 1) * 64);

        v8i afr[2];
#pragma unroll
        for (int sub = 0; sub < 2; sub++) {
            const signed char* ap =
                &lsA[cur][(wave * 32 + sub * 16 + m_l) * 64 + half * 8];
            unsigned long long a01 = *(const unsigned long long*)(ap);
            unsigned long long a23 = *(const unsigned long long*)(ap + 16);
            unsigned long long a45 = *(const unsigned long long*)(ap + 32);
            unsigned long long a67 = *(const unsigned long long*)(ap + 48);
            afr[sub] = (v8i){(int)(unsigned)a01, (int)(a01 >> 32),
                             (int)(unsigned)a23, (int)(a23 >> 32),
                             (int)(unsigned)a45, (int)(a45 >> 32),
                             (int)(unsigned)a67, (int)(a67 >> 32)};
        }

#pragma unroll
        for (int nt = 0; nt < 4; nt++) {
            const signed char* bp = &lsB[cur][(nt * 16 + m_l) * 64 + half * 8];
            int2 b0 = *(const int2*)(bp);
            int2 b1 = *(const int2*)(bp + 16);
            int2 b2 = *(const int2*)(bp + 32);
            int2 b3 = *(const int2*)(bp + 48);
            v8i b = {b0.x, b0.y, b1.x, b1.y, b2.x, b2.y, b3.x, b3.y};
            acc[0][nt] = __builtin_amdgcn_wmma_i32_16x16x64_iu8(
                true, afr[0], true, b, acc[0][nt], false, false);
            acc[1][nt] = __builtin_amdgcn_wmma_i32_16x16x64_iu8(
                true, afr[1], true, b, acc[1][nt], false, false);
        }
    }

    float wdeq = *wdeq_ptr;
#pragma unroll
    for (int sub = 0; sub < 2; sub++) {
        float rd[8];
#pragma unroll
        for (int v = 0; v < 8; v++)
            rd[v] = row_deq[m0 + wave * 32 + sub * 16 + half * 8 + v];
#pragma unroll
        for (int nt = 0; nt < 4; nt++) {
#pragma unroll
            for (int v = 0; v < 8; v++) {
                float val = (float)acc[sub][nt][v] * rd[v] * wdeq;
                if (mode == 1) val = 1.0f / (1.0f + __expf(-val));
                size_t mg = (size_t)(m0 + wave * 32 + sub * 16 + half * 8 + v);
                C[mg * Ndim + (n0 + nt * 16 + m_l)] = val;
            }
        }
    }
}


__global__ void __launch_bounds__(256)
wkv_phase1(const float* __restrict__ td,
           const float* __restrict__ k, const float* __restrict__ v,
           float* __restrict__ a_loc, float* __restrict__ b_loc,
           float* __restrict__ e_loc) {
    int gid = blockIdx.x * 256 + threadIdx.x;
    int c  = gid >> 13;
    int ch = gid & (NCH - 1);
    int b  = ch >> 10;
    int d  = ch & (Dz - 1);
    float wlog = -__expf(td[d]);
    float alpha = 0.f, beta = 0.f, eps = 0.f;
    size_t idx = ((size_t)b * Tz + (size_t)c * CHL) * Dz + d;
    for (int t = 0; t < CHL; t++, idx += Dz) {
        float kt = k[idx];
        float vt = v[idx];
        float weps = eps + wlog;
        float epsn = fmaxf(weps, kt);
        float e1n = __expf(weps - epsn);
        float e2n = __expf(kt - epsn);
        alpha = e1n * alpha + e2n * vt;
        beta  = e1n * beta + e2n;
        eps   = epsn;
    }
    a_loc[gid] = alpha;
    b_loc[gid] = beta;
    e_loc[gid] = eps;
}

__global__ void __launch_bounds__(256)
wkv_scan(const float* __restrict__ td,
         const float* __restrict__ a_loc, const float* __restrict__ b_loc,
         const float* __restrict__ e_loc,
         float* __restrict__ a_in, float* __restrict__ b_in,
         float* __restrict__ e_in) {
    int ch = blockIdx.x * 256 + threadIdx.x;
    int d = ch & (Dz - 1);
    float wL = -__expf(td[d]) * (float)CHL;
    float A = 0.f, B = 0.f, E = 0.f;
    for (int j = 0; j < CHK; j++) {
        int o = j * NCH + ch;
        a_in[o] = A; b_in[o] = B; e_in[o] = E;
        float Ed = E + wL;
        float al = a_loc[o], bl = b_loc[o], el = e_loc[o];
        float m = fmaxf(Ed, el);
        float x1 = __expf(Ed - m);
        float x2 = __expf(el - m);
        A = x1 * A + x2 * al;
        B = x1 * B + x2 * bl;
        E = m;
    }
}

__global__ void __launch_bounds__(256)
wkv_phase3(const float* __restrict__ td, const float* __restrict__ tf,
           const float* __restrict__ k, const float* __restrict__ v,
           const float* __restrict__ r,
           const float* __restrict__ a_in, const float* __restrict__ b_in,
           const float* __restrict__ e_in, float* __restrict__ y) {
    int gid = blockIdx.x * 256 + threadIdx.x;
    int c  = gid >> 13;
    int ch = gid & (NCH - 1);
    int b  = ch >> 10;
    int d  = ch & (Dz - 1);
    float wlog = -__expf(td[d]);
    float u = tf[d];
    float alpha = a_in[gid], beta = b_in[gid], eps = e_in[gid];
    size_t idx = ((size_t)b * Tz + (size_t)c * CHL) * Dz + d;
    for (int t = 0; t < CHL; t++, idx += Dz) {
        float kt = k[idx];
        float vt = v[idx];
        float ukt = u + kt;
        float tau = fmaxf(ukt, eps);
        float e1 = __expf(eps - tau);
        float e2 = __expf(ukt - tau);
        float out = (e1 * alpha + e2 * vt) / (e1 * beta + e2);
        float weps = eps + wlog;
        float epsn = fmaxf(weps, kt);
        float e1n = __expf(weps - epsn);
        float e2n = __expf(kt - epsn);
        alpha = e1n * alpha + e2n * vt;
        beta  = e1n * beta + e2n;
        eps   = epsn;
        y[idx] = r[idx] * out;
    }
}

extern "C" void kernel_launch(void* const* d_in, const int* in_sizes, int n_in,
                              void* d_out, int out_size, void* d_ws, size_t ws_size,
                              hipStream_t stream) {
    const float* hidden = (const float*)d_in[0];
    const float* tdec   = (const float*)d_in[1];
    const float* tfst   = (const float*)d_in[2];
    const float* tmk    = (const float*)d_in[3];
    const float* tmv    = (const float*)d_in[4];
    const float* tmr    = (const float*)d_in[5];
    const float* wk     = (const float*)d_in[6];
    const float* wv     = (const float*)d_in[7];
    const float* wr     = (const float*)d_in[8];
    const float* wo     = (const float*)d_in[9];
    const float* rmsk   = (const float*)d_in[10];
    const float* rmsv   = (const float*)d_in[11];
    const float* rmsr   = (const float*)d_in[12];
    const float* rmso   = (const float*)d_in[13];
    float* out = (float*)d_out;

    char* ws = (char*)d_ws;
    const size_t MB = 1ull << 20;
    float* wdeq     = (float*)(ws + 32);
    float* partials = (float*)(ws + 64);
    signed char* wq[4];
    for (int i = 0; i < 4; i++) wq[i] = (signed char*)(ws + 4096 + (size_t)i * MB);
    signed char* xqk = (signed char*)(ws + 8 * MB);
    signed char* xqv = (signed char*)(ws + 24 * MB);
    signed char* xqr = (signed char*)(ws + 40 * MB);
    signed char* xqo = (signed char*)(ws + 56 * MB);
    float* rdk = (float*)(ws + 72 * MB);
    float* rdv = (float*)(ws + 72 * MB + 64 * 1024);
    float* rdr = (float*)(ws + 72 * MB + 128 * 1024);
    float* rdo = (float*)(ws + 72 * MB + 192 * 1024);
    float* kbuf = (float*)(ws + 73 * MB);
    float* vbuf = (float*)(ws + 137 * MB);
    float* rbuf = (float*)(ws + 201 * MB);
    float* ybuf = (float*)(ws + 265 * MB);
    float* a_loc = (float*)(ws + 330 * MB);
    float* b_loc = (float*)(ws + 331 * MB);
    float* e_loc = (float*)(ws + 332 * MB);
    float* a_in  = (float*)(ws + 333 * MB);
    float* b_in  = (float*)(ws + 334 * MB);
    float* e_in  = (float*)(ws + 335 * MB);

    w_abssum<<<dim3(64, 4), 256, 0, stream>>>(wk, wv, wr, wo, partials);
    w_finalize<<<4, 64, 0, stream>>>(partials, wdeq);
    w_quant<<<dim3(1024, 4), 256, 0, stream>>>(wk, wv, wr, wo, wdeq,
                                               wq[0], wq[1], wq[2], wq[3]);

    mix_rms_quant<<<Mtok, 256, 0, stream>>>(hidden, tmk, tmv, tmr,
                                            rmsk, rmsv, rmsr,
                                            xqk, xqv, xqr, rdk, rdv, rdr);

    dim3 ggrid(Mtok / 256, Ndim / 64);
    gemm_i8<<<ggrid, 256, 0, stream>>>(xqk, wq[0], rdk, wdeq + 0, kbuf, 0);
    gemm_i8<<<ggrid, 256, 0, stream>>>(xqv, wq[1], rdv, wdeq + 1, vbuf, 0);
    gemm_i8<<<ggrid, 256, 0, stream>>>(xqr, wq[2], rdr, wdeq + 2, rbuf, 1);

    wkv_phase1<<<(CHK * NCH) / 256, 256, 0, stream>>>(tdec, kbuf, vbuf,
                                                      a_loc, b_loc, e_loc);
    wkv_scan<<<NCH / 256, 256, 0, stream>>>(tdec, a_loc, b_loc, e_loc,
                                            a_in, b_in, e_in);
    wkv_phase3<<<(CHK * NCH) / 256, 256, 0, stream>>>(tdec, tfst, kbuf, vbuf,
                                                      rbuf, a_in, b_in, e_in,
                                                      ybuf);

    rms_quant_single<<<Mtok, 256, 0, stream>>>(ybuf, rmso, xqo, rdo);
    gemm_i8<<<ggrid, 256, 0, stream>>>(xqo, wq[3], rdo, wdeq + 3, out, 0);

    (void)in_sizes; (void)n_in; (void)out_size; (void)ws_size;
}
